// CrossAttention_Mirror_38663295598641
// MI455X (gfx1250) — matrix operand order repaired and checked by reading, not compiled, not run on hardware
//
#include <hip/hip_runtime.h>

// ---------------------------------------------------------------------------
// MI455X (gfx1250, wave32) implementation of the cross/self-attention head.
// Heavy math on v_wmma_f32_16x16x32_f16; flash-style attention (no 268MB
// [B,N,N] materialization); 3x3 convs as 9 shifted GEMM taps; the 258-channel
// self-attention path zero-padded to 288 (=9*32).
// This round: double-buffered TDM pipeline in k_gemm/k_conv3 — the Tensor
// Data Mover prefetches the next A tile and the next B tile's global loads
// sit in registers while the current tile's WMMAs run; s_wait_tensorcnt<=1
// exploits in-order TENSORcnt completion.
// ---------------------------------------------------------------------------

#ifndef __has_builtin
#define __has_builtin(x) 0
#endif

#if __has_builtin(__builtin_amdgcn_tensor_load_to_lds)
#define HAVE_TDM 1
#else
#define HAVE_TDM 0
#endif

#if defined(__has_include)
#if __has_include(<hip/amd_detail/amd_gfx1250_TDM.h>)
#define TDM_6ARG 1      // therock-10.0 headers -> clang-23 6-arg builtin
#else
#define TDM_6ARG 0      // ROCm 7.2 clang-22 5-arg builtin
#endif
#else
#define TDM_6ARG 0
#endif

typedef __attribute__((ext_vector_type(16))) _Float16 v16h;
typedef __attribute__((ext_vector_type(8)))  _Float16 v8h;
typedef __attribute__((ext_vector_type(8)))  float    v8f;
typedef unsigned int v4u __attribute__((ext_vector_type(4)));
typedef int          v4i __attribute__((ext_vector_type(4)));
typedef int          v8i __attribute__((ext_vector_type(8)));

#define NSP 4096      // H*W
#define IMG_W 64

static __device__ __forceinline__ v8f wmma_f16(v16h a, v16h b, v8f c) {
  return __builtin_amdgcn_wmma_f32_16x16x32_f16(false, a, false, b,
                                                (short)0, c, false, false);
}

// two aligned 16B LDS reads composed into one 16xf16 fragment
static __device__ __forceinline__ v16h ld_pair(const _Float16* p0,
                                               const _Float16* p1) {
  v8h a = *(const v8h*)p0;
  v8h b = *(const v8h*)p1;
  return __builtin_shufflevector(a, b, 0,1,2,3,4,5,6,7,8,9,10,11,12,13,14,15);
}

// build f16 fragment from two runs of 8 contiguous f32 in LDS
static __device__ __forceinline__ v16h frag_from_f32(const float* p0,
                                                     const float* p1) {
  v8f a = *(const v8f*)p0;
  v8f b = *(const v8f*)p1;
  v16h r;
#pragma unroll
  for (int j = 0; j < 8; ++j) { r[j] = (_Float16)a[j]; r[8 + j] = (_Float16)b[j]; }
  return r;
}

// ---------------------------------------------------------------------------
// Tensor Data Mover: DMA a 2D f32 tile (tile_w x tile_h elems, row stride
// stride_elems) from global memory into LDS at byte offset lds_off.
// D# packing per CDNA5 ISA (group0 128b, group1 256b;
// groups 2/3 zero for a 2D tensor; tensor dims == tile dims, fully in-bounds).
// ---------------------------------------------------------------------------
#if HAVE_TDM
static __device__ __forceinline__ void tdm_load_2d(unsigned lds_off,
                                                   const float* gsrc,
                                                   unsigned tile_w,
                                                   unsigned tile_h,
                                                   unsigned stride_elems) {
  unsigned long long ga = (unsigned long long)(void*)gsrc;
  v4u g0;
  g0[0] = 1u;                                            // count=1, user D#
  g0[1] = lds_off;                                       // lds_addr (bytes)
  g0[2] = (unsigned)(ga & 0xFFFFFFFFu);                  // global_addr[31:0]
  g0[3] = (unsigned)((ga >> 32) & 0x01FFFFFFu) | (2u << 30); // addr[56:32]|type=2
  v8i g1;
  g1[0] = (int)(2u << 16);                               // data_size=2 -> 4B
  g1[1] = (int)((tile_w & 0xFFFFu) << 16);               // tensor_dim0[15:0]
  g1[2] = (int)((tile_h & 0xFFFFu) << 16);               // tensor_dim1[15:0]
  g1[3] = (int)(tile_w << 16);                           // tile_dim0
  g1[4] = (int)(tile_h & 0xFFFFu);                       // tile_dim1 (dim2=0)
  g1[5] = (int)stride_elems;                             // dim0_stride[31:0]
  g1[6] = 0;
  g1[7] = 0;
  v4i g2 = {0, 0, 0, 0};
  v4i g3 = {0, 0, 0, 0};
#if TDM_6ARG
  v8i g4 = {0, 0, 0, 0, 0, 0, 0, 0};
  __builtin_amdgcn_tensor_load_to_lds(g0, g1, g2, g3, g4, 0);
#else
  __builtin_amdgcn_tensor_load_to_lds(g0, g1, g2, g3, 0);
#endif
}
static __device__ __forceinline__ void tdm_wait_le0() {
#if __has_builtin(__builtin_amdgcn_s_wait_tensorcnt)
  __builtin_amdgcn_s_wait_tensorcnt(0);
#endif
}
static __device__ __forceinline__ void tdm_wait_le1() {
#if __has_builtin(__builtin_amdgcn_s_wait_tensorcnt)
  __builtin_amdgcn_s_wait_tensorcnt(1);
#endif
}
#endif  // HAVE_TDM

// ---------------------------------------------------------------------------
// Generic GEMM: D[m,n] = sum_k A[m,k] * X[k,n] (+bias[m])
// A row-major [M,K] (lda=K), X channel-major [K,4096], D [M,4096].
// Block = 128 threads (4 waves), tile 64(M) x 64(N), K step 32.
// Double-buffered pipeline: TDM DMAs A(s+1), B(s+1) global loads sit in
// registers across the WMMAs of step s; one barrier pair per step.
// ---------------------------------------------------------------------------
__global__ __launch_bounds__(128, 1)
void k_gemm(const float* __restrict__ A, int lda,
            const float* __restrict__ X, long long xBatch,
            float* __restrict__ D, long long dBatch,
            int K, int Mvalid, const float* __restrict__ bias)
{
  __shared__ float    lAf[2][64][32];  // [m][k] f32 (TDM destination)
  __shared__ _Float16 lB[2][64][32];   // [n][k] f16 (transposed stage)
  const int tid   = threadIdx.x;
  const int wave  = tid >> 5;
  const int lane  = tid & 31;
  const int lhalf = lane >> 4, l15 = lane & 15;
  const int m0 = blockIdx.y * 64;
  const int n0 = blockIdx.x * 64;
  const float* Xb = X + (long long)blockIdx.z * xBatch;
  float*       Db = D + (long long)blockIdx.z * dBatch;
#if HAVE_TDM
  const unsigned ldsA = (unsigned)(unsigned long long)(void*)&lAf[0][0][0];
#endif

  v8f acc[4];
#pragma unroll
  for (int t = 0; t < 4; ++t)
#pragma unroll
    for (int v = 0; v < 8; ++v) acc[t][v] = 0.0f;

  const int am = tid >> 1, ak = (tid & 1) * 16;   // fallback A stage
  const int bk = tid >> 2, bn = (tid & 3) * 16;   // B stage: 16 contiguous n
  (void)am; (void)ak;
  const int steps = K >> 5;

  auto stageA = [&](int s, int buf) {
    const float* asrc = A + (long long)m0 * lda + (s << 5);
#if HAVE_TDM
    if (tid < 32)
      tdm_load_2d(ldsA + (unsigned)buf * (64 * 32 * 4), asrc, 32, 64,
                  (unsigned)lda);
#else
    const float* as = asrc + (long long)am * lda + ak;
#pragma unroll
    for (int j = 0; j < 16; ++j) lAf[buf][am][ak + j] = as[j];
#endif
  };

  // prologue: tile 0 fully staged
  stageA(0, 0);
  {
    const float* bs = Xb + (long long)bk * NSP + n0 + bn;
#pragma unroll
    for (int j = 0; j < 16; ++j) lB[0][bn + j][bk] = (_Float16)bs[j];
  }

  for (int s = 0; s < steps; ++s) {
    const int  cur  = s & 1, nxt = cur ^ 1;
    const bool more = (s + 1 < steps);
    float breg[16];
    if (more) {
      stageA(s + 1, nxt);   // async DMA overlaps this step's math
      const float* bs = Xb + (long long)(((s + 1) << 5) + bk) * NSP + n0 + bn;
#pragma unroll
      for (int j = 0; j < 16; ++j) breg[j] = bs[j];     // loads stay in flight
    }
#if HAVE_TDM
    if (tid < 32) { if (more) tdm_wait_le1(); else tdm_wait_le0(); }
#endif
    __syncthreads();
    const float* pa = &lAf[cur][wave * 16 + l15][lhalf * 8];
    v16h af = frag_from_f32(pa, pa + 16);
#pragma unroll
    for (int t = 0; t < 4; ++t) {
      const _Float16* pb = &lB[cur][t * 16 + l15][lhalf * 8];
      acc[t] = wmma_f16(af, ld_pair(pb, pb + 16), acc[t]);
    }
    if (more) {
#pragma unroll
      for (int j = 0; j < 16; ++j) lB[nxt][bn + j][bk] = (_Float16)breg[j];
    }
    __syncthreads();   // WAR: next TDM/store may not touch bufs until all read
  }

#pragma unroll
  for (int t = 0; t < 4; ++t)
#pragma unroll
    for (int v = 0; v < 8; ++v) {
      const int m = m0 + wave * 16 + 8 * lhalf + v;
      if (m < Mvalid) {
        const int n = n0 + t * 16 + l15;
        float r = acc[t][v];
        if (bias) r += bias[m];
        Db[(long long)m * NSP + n] = r;
      }
    }
}

// ---------------------------------------------------------------------------
// 3x3 conv as 9 shifted GEMM taps, flattened into one pipelined step stream
// (step = tap * (Kc/32) + kchunk) so accumulators stay resident and the TDM
// double-buffer never drains. Wp layout [tap][256][Kc] (zero padded).
// Block tile: 64 out-channels x one image row (64 pixels).
// ---------------------------------------------------------------------------
__global__ __launch_bounds__(128, 1)
void k_conv3(const float* __restrict__ Wp, const float* __restrict__ X,
             long long xBatch, int Kc, float* __restrict__ D)
{
  __shared__ float    lAf[2][64][32];
  __shared__ _Float16 lB[2][64][32];
  const int tid   = threadIdx.x;
  const int wave  = tid >> 5;
  const int lane  = tid & 31;
  const int lhalf = lane >> 4, l15 = lane & 15;
  const int h  = blockIdx.x;          // image row == N tile
  const int m0 = blockIdx.y * 64;
  const float* Xb = X + (long long)blockIdx.z * xBatch;
  float*       Db = D + (long long)blockIdx.z * (256LL * NSP);
#if HAVE_TDM
  const unsigned ldsA = (unsigned)(unsigned long long)(void*)&lAf[0][0][0];
#endif

  v8f acc[4];
#pragma unroll
  for (int t = 0; t < 4; ++t)
#pragma unroll
    for (int v = 0; v < 8; ++v) acc[t][v] = 0.0f;

  const int am = tid >> 1, ak = (tid & 1) * 16;
  const int bk = tid >> 2, bw = (tid & 3) * 16;
  (void)am; (void)ak;
  const int kch   = Kc >> 5;
  const int steps = 9 * kch;

  auto stageA = [&](int s, int buf) {
    const int tap = s / kch, kc = s - tap * kch;
    const float* asrc =
        Wp + (long long)tap * 256 * Kc + (long long)m0 * Kc + (kc << 5);
#if HAVE_TDM
    if (tid < 32)
      tdm_load_2d(ldsA + (unsigned)buf * (64 * 32 * 4), asrc, 32, 64,
                  (unsigned)Kc);
#else
    const float* as = asrc + (long long)am * Kc + ak;
#pragma unroll
    for (int j = 0; j < 16; ++j) lAf[buf][am][ak + j] = as[j];
#endif
  };
  auto loadB = [&](int s, float* breg) {
    const int tap = s / kch, kc = s - tap * kch;
    const int dy = tap / 3 - 1, dx = tap % 3 - 1;
    const int hh = h + dy;
    const bool hv = (hh >= 0) && (hh < IMG_W);
    const float* bs = Xb + (long long)((kc << 5) + bk) * NSP + hh * IMG_W;
#pragma unroll
    for (int j = 0; j < 16; ++j) {
      const int w = bw + j + dx;
      breg[j] = (hv && w >= 0 && w < IMG_W) ? bs[w] : 0.0f;
    }
  };

  // prologue: step 0 fully staged
  stageA(0, 0);
  {
    float b0[16];
    loadB(0, b0);
#pragma unroll
    for (int j = 0; j < 16; ++j) lB[0][bw + j][bk] = (_Float16)b0[j];
  }

  for (int s = 0; s < steps; ++s) {
    const int  cur  = s & 1, nxt = cur ^ 1;
    const bool more = (s + 1 < steps);
    float breg[16];
    if (more) {
      stageA(s + 1, nxt);
      loadB(s + 1, breg);
    }
#if HAVE_TDM
    if (tid < 32) { if (more) tdm_wait_le1(); else tdm_wait_le0(); }
#endif
    __syncthreads();
    const float* pa = &lAf[cur][wave * 16 + l15][lhalf * 8];
    v16h af = frag_from_f32(pa, pa + 16);
#pragma unroll
    for (int t = 0; t < 4; ++t) {
      const _Float16* pb = &lB[cur][t * 16 + l15][lhalf * 8];
      acc[t] = wmma_f16(af, ld_pair(pb, pb + 16), acc[t]);
    }
    if (more) {
#pragma unroll
      for (int j = 0; j < 16; ++j) lB[nxt][bw + j][bk] = (_Float16)breg[j];
    }
    __syncthreads();
  }

#pragma unroll
  for (int t = 0; t < 4; ++t)
#pragma unroll
    for (int v = 0; v < 8; ++v) {
      const int m = m0 + wave * 16 + 8 * lhalf + v;
      const int n = h * IMG_W + t * 16 + l15;
      Db[(long long)m * NSP + n] = acc[t][v];
    }
}

// ---------------------------------------------------------------------------
// Flash attention (faithful: no 1/sqrt(d) scale):
//   S[q,m] = sum_c Q[c,q]*K[c,m];  P = softmax_m(S);  O[c,q] = sum_m V[c,m]*P[q,m]
//   out = gamma*O + R   (R = residual, channel-major [CP,4096])
// Block: 128 threads, 64-query tile, streams 32-key tiles through LDS.
// Q fragments in registers (KC v16h/lane); O accumulators (CT v8f) resident
// in VGPRs with online max/sum rescaling.
// ---------------------------------------------------------------------------
template<int CP>
__global__ __launch_bounds__(128, 1)
void k_flash(const float* __restrict__ Q, const float* __restrict__ Kt,
             const float* __restrict__ V, const float* __restrict__ R,
             float* __restrict__ O, const float* __restrict__ gamma)
{
  constexpr int CT = CP / 16;   // output channel tiles
  constexpr int KC = CP / 32;   // k-chunks for S gemm
  __shared__ _Float16 lK[32][CP];   // [m][c]
  __shared__ _Float16 lV[CP][32];   // [c][m]
  __shared__ _Float16 lP[64][32];   // probabilities (A operand of O-gemm)
  __shared__ float    lS[64][32];
  __shared__ float    lM[64], lL[64], lScale[64];

  const int tid   = threadIdx.x;
  const int wave  = tid >> 5;
  const int lane  = tid & 31;
  const int lhalf = lane >> 4, l15 = lane & 15;
  const int n0 = blockIdx.x * 64;
  const long long bs = (long long)CP * NSP;
  const float* Qb = Q + blockIdx.z * bs;
  const float* Kb = Kt + blockIdx.z * bs;
  const float* Vb = V + blockIdx.z * bs;
  const float* Rb = R + blockIdx.z * bs;
  float*       Ob = O + blockIdx.z * bs;
  const int srow = wave * 16 + 8 * lhalf;   // first D row held by this lane

  // Q fragments, A layout: row = l15 (query), K = channel
  v16h qf[KC];
  const int qn = n0 + wave * 16 + l15;
#pragma unroll
  for (int kc = 0; kc < KC; ++kc) {
    const int c0 = kc * 32 + 8 * lhalf;
#pragma unroll
    for (int j = 0; j < 8; ++j) {
      qf[kc][j]     = (_Float16)Qb[(long long)(c0 + j) * NSP + qn];
      qf[kc][8 + j] = (_Float16)Qb[(long long)(c0 + 16 + j) * NSP + qn];
    }
  }

  v8f oacc[CT];
#pragma unroll
  for (int t = 0; t < CT; ++t)
#pragma unroll
    for (int v = 0; v < 8; ++v) oacc[t][v] = 0.0f;

  if (tid < 64) { lM[tid] = -3.0e38f; lL[tid] = 0.0f; }

  for (int m0 = 0; m0 < NSP; m0 += 32) {
    __syncthreads();
    // stage K (transposed) and V tiles
    for (int c = tid; c < CP; c += 128) {
      const float* ks = Kb + (long long)c * NSP + m0;
      const float* vs = Vb + (long long)c * NSP + m0;
#pragma unroll
      for (int j = 0; j < 32; ++j) {
        lK[j][c] = (_Float16)ks[j];
        lV[c][j] = (_Float16)vs[j];
      }
    }
    __syncthreads();

    // S tile: 16 queries (this wave) x 32 keys
    v8f s0, s1;
#pragma unroll
    for (int v = 0; v < 8; ++v) { s0[v] = 0.0f; s1[v] = 0.0f; }
#pragma unroll
    for (int kc = 0; kc < KC; ++kc) {
      const int c0 = kc * 32 + lhalf * 8;
      const _Float16* pb0 = &lK[l15][c0];
      const _Float16* pb1 = &lK[16 + l15][c0];
      s0 = wmma_f16(qf[kc], ld_pair(pb0, pb0 + 16), s0);
      s1 = wmma_f16(qf[kc], ld_pair(pb1, pb1 + 16), s1);
    }
#pragma unroll
    for (int v = 0; v < 8; ++v) {
      lS[srow + v][l15]      = s0[v];
      lS[srow + v][16 + l15] = s1[v];
    }
    __syncthreads();

    // online softmax per query row
    if (tid < 64) {
      const float mOld = lM[tid];
      float mx = mOld;
      float sv[32];
#pragma unroll
      for (int j = 0; j < 32; ++j) { sv[j] = lS[tid][j]; mx = fmaxf(mx, sv[j]); }
      const float sc = __expf(mOld - mx);
      float sum = 0.0f;
#pragma unroll
      for (int j = 0; j < 32; ++j) {
        const float p = __expf(sv[j] - mx);
        sum += p;
        lP[tid][j] = (_Float16)p;
      }
      lL[tid] = lL[tid] * sc + sum;
      lM[tid] = mx;
      lScale[tid] = sc;
    }
    __syncthreads();

    // rescale + accumulate O += P * V^T
    float scl[8];
#pragma unroll
    for (int v = 0; v < 8; ++v) scl[v] = lScale[srow + v];
    const _Float16* pp = &lP[wave * 16 + l15][lhalf * 8];
    v16h pf = ld_pair(pp, pp + 16);
#pragma unroll
    for (int t = 0; t < CT; ++t) {
#pragma unroll
      for (int v = 0; v < 8; ++v) oacc[t][v] *= scl[v];
      const _Float16* pv = &lV[t * 16 + l15][lhalf * 8];
      oacc[t] = wmma_f16(pf, ld_pair(pv, pv + 16), oacc[t]);
    }
  }

  __syncthreads();
  const float g = gamma[0];
  float rl[8];
#pragma unroll
  for (int v = 0; v < 8; ++v) rl[v] = 1.0f / lL[srow + v];
#pragma unroll
  for (int t = 0; t < CT; ++t)
#pragma unroll
    for (int v = 0; v < 8; ++v) {
      const int c = t * 16 + l15;
      const int n = n0 + srow + v;
      const long long off = (long long)c * NSP + n;
      Ob[off] = g * (oacc[t][v] * rl[v]) + Rb[off];
    }
}

// ---------------------------------------------------------------------------
// BatchNorm (training-mode biased stats) + ReLU, layout [4][256][4096]
// ---------------------------------------------------------------------------
__global__ void k_bnstats(const float* __restrict__ X, float* __restrict__ mean,
                          float* __restrict__ istd)
{
  __shared__ float s1[256], s2[256];
  const int c = blockIdx.x, tid = threadIdx.x;
  float sum = 0.0f, sq = 0.0f;
  for (int i = tid; i < 4 * NSP; i += 256) {
    const int b = i >> 12, n = i & (NSP - 1);
    const float v = X[(long long)b * 256 * NSP + (long long)c * NSP + n];
    sum += v; sq += v * v;
  }
  s1[tid] = sum; s2[tid] = sq; __syncthreads();
  for (int st = 128; st > 0; st >>= 1) {
    if (tid < st) { s1[tid] += s1[tid + st]; s2[tid] += s2[tid + st]; }
    __syncthreads();
  }
  if (tid == 0) {
    const float m   = s1[0] / (4.0f * NSP);
    const float var = s2[0] / (4.0f * NSP) - m * m;
    mean[c] = m;
    istd[c] = rsqrtf(var + 1e-5f);
  }
}

__global__ void k_bnrelu(float* __restrict__ X, const float* __restrict__ mean,
                         const float* __restrict__ istd, const float* __restrict__ g,
                         const float* __restrict__ bb)
{
  const long long idx = (long long)blockIdx.x * blockDim.x + threadIdx.x;
  const int c = (int)((idx >> 12) & 255);
  float v = X[idx];
  v = (v - mean[c]) * istd[c] * g[c] + bb[c];
  X[idx] = v > 0.0f ? v : 0.0f;
}

// cat: [cross(256) | flow(2) | zeros(30)] -> [4][288][4096]
__global__ void k_cat(const float* __restrict__ cross, const float* __restrict__ flow,
                      float* __restrict__ out)
{
  const long long idx = (long long)blockIdx.x * blockDim.x + threadIdx.x;
  const int n = (int)(idx & (NSP - 1));
  const int c = (int)((idx >> 12) % 288);
  const int b = (int)(idx / (288LL * NSP));
  float v = 0.0f;
  if (c < 256)      v = cross[(long long)b * 256 * NSP + (long long)c * NSP + n];
  else if (c < 258) v = flow[(long long)b * 2 * NSP + (long long)(c - 256) * NSP + n];
  out[idx] = v;
}

// zero-pad weight [Min,Kin] -> [Mout,Kout]
__global__ void k_padw(const float* __restrict__ in, float* __restrict__ out,
                       int Min, int Kin, int Mout, int Kout)
{
  const int idx = blockIdx.x * blockDim.x + threadIdx.x;
  if (idx >= Mout * Kout) return;
  const int o = idx / Kout, i = idx - o * Kout;
  out[idx] = (o < Min && i < Kin) ? in[o * Kin + i] : 0.0f;
}

// repack conv weight [O,Cin,3,3] -> [tap][256][Kc] zero padded in C
__global__ void k_repack(const float* __restrict__ w, float* __restrict__ out,
                         int Cin, int Kc)
{
  const int idx = blockIdx.x * blockDim.x + threadIdx.x;
  if (idx >= 9 * 256 * Kc) return;
  const int tap = idx / (256 * Kc);
  const int rem = idx - tap * 256 * Kc;
  const int o = rem / Kc, c = rem - o * Kc;
  out[idx] = (c < Cin) ? w[((o * Cin) + c) * 9 + tap] : 0.0f;
}

// pred: out[b,n] = sum_c w[c]*X[b,c,n] + bias
__global__ void k_pred(const float* __restrict__ X, const float* __restrict__ w,
                       const float* __restrict__ bias, float* __restrict__ out)
{
  const int idx = blockIdx.x * blockDim.x + threadIdx.x;
  const int b = idx >> 12, n = idx & (NSP - 1);
  float s = bias[0];
#pragma unroll 4
  for (int c = 0; c < 256; ++c)
    s += w[c] * X[(long long)b * 256 * NSP + (long long)c * NSP + n];
  out[idx] = s;
}

// ---------------------------------------------------------------------------
extern "C" void kernel_launch(void* const* d_in, const int* in_sizes, int n_in,
                              void* d_out, int out_size, void* d_ws, size_t ws_size,
                              hipStream_t stream)
{
  (void)in_sizes; (void)n_in; (void)out_size; (void)ws_size;
  const float* x     = (const float*)d_in[0];
  const float* y     = (const float*)d_in[1];
  const float* flow  = (const float*)d_in[2];
  const float* w1x1  = (const float*)d_in[3];
  const float* b1x1  = (const float*)d_in[4];
  const float* ca_wq = (const float*)d_in[5];
  const float* ca_wk = (const float*)d_in[6];
  const float* ca_wv = (const float*)d_in[7];
  const float* ca_g  = (const float*)d_in[8];
  const float* cbr1  = (const float*)d_in[9];
  const float* bn1g  = (const float*)d_in[10];
  const float* bn1b  = (const float*)d_in[11];
  const float* sa_wq = (const float*)d_in[12];
  const float* sa_wk = (const float*)d_in[13];
  const float* sa_wv = (const float*)d_in[14];
  const float* sa_g  = (const float*)d_in[15];
  const float* cbr2  = (const float*)d_in[16];
  const float* bn2g  = (const float*)d_in[17];
  const float* bn2b  = (const float*)d_in[18];
  const float* predw = (const float*)d_in[19];
  const float* predb = (const float*)d_in[20];
  float* out = (float*)d_out;

  // workspace layout (floats): six 288-channel activation buffers + weights
  float* ws = (float*)d_ws;
  const long long BIG = 4LL * 288 * NSP;
  float* B0 = ws;             // xb  -> q2
  float* B1 = B0 + BIG;       // yb  -> k2
  float* B2 = B1 + BIG;       // q   -> conv1 out/cross -> v2
  float* B3 = B2 + BIG;       // k   -> cat
  float* B4 = B3 + BIG;       // v   -> t2
  float* B5 = B4 + BIG;       // t1  -> conv2 out
  float* Wq2 = B5 + BIG;      // padded self-attn weights [320][288]
  float* Wk2 = Wq2 + 320 * 288;
  float* Wv2 = Wk2 + 320 * 288;
  float* Wp1 = Wv2 + 320 * 288;          // [9][256][256]
  float* Wp2 = Wp1 + 9 * 256 * 256;      // [9][256][288]
  float* st  = Wp2 + 9 * 256 * 288;      // mean1,istd1,mean2,istd2

  const long long XB512 = 512LL * NSP, XB256 = 256LL * NSP, XB288 = 288LL * NSP;
  dim3 blk(128);

  // weight prep
  k_padw<<<(320 * 288 + 255) / 256, 256, 0, stream>>>(sa_wq, Wq2, 258, 258, 320, 288);
  k_padw<<<(320 * 288 + 255) / 256, 256, 0, stream>>>(sa_wk, Wk2, 258, 258, 320, 288);
  k_padw<<<(320 * 288 + 255) / 256, 256, 0, stream>>>(sa_wv, Wv2, 258, 258, 320, 288);
  k_repack<<<(9 * 256 * 256 + 255) / 256, 256, 0, stream>>>(cbr1, Wp1, 256, 256);
  k_repack<<<(9 * 256 * 288 + 255) / 256, 256, 0, stream>>>(cbr2, Wp2, 258, 288);

  // bottleneck 1x1 convs
  k_gemm<<<dim3(64, 4, 4), blk, 0, stream>>>(w1x1, 512, x, XB512, B0, XB256, 512, 256, b1x1);
  k_gemm<<<dim3(64, 4, 4), blk, 0, stream>>>(w1x1, 512, y, XB512, B1, XB256, 512, 256, b1x1);
  // cross-attention qkv
  k_gemm<<<dim3(64, 4, 4), blk, 0, stream>>>(ca_wq, 256, B0, XB256, B2, XB256, 256, 256, nullptr);
  k_gemm<<<dim3(64, 4, 4), blk, 0, stream>>>(ca_wk, 256, B1, XB256, B3, XB256, 256, 256, nullptr);
  k_gemm<<<dim3(64, 4, 4), blk, 0, stream>>>(ca_wv, 256, B1, XB256, B4, XB256, 256, 256, nullptr);
  // flash cross-attention (+gamma residual with xb)
  k_flash<256><<<dim3(64, 1, 4), blk, 0, stream>>>(B2, B3, B4, B0, B5, ca_g);
  // conv3x3 #1 + BN + ReLU
  k_conv3<<<dim3(64, 4, 4), blk, 0, stream>>>(Wp1, B5, XB256, 256, B2);
  k_bnstats<<<256, 256, 0, stream>>>(B2, st, st + 256);
  k_bnrelu<<<16384, 256, 0, stream>>>(B2, st, st + 256, bn1g, bn1b);
  // concat with flow, zero-pad to 288 channels
  k_cat<<<(int)(BIG / 256), 256, 0, stream>>>(B2, flow, B3);
  // self-attention qkv (padded 288-dim)
  k_gemm<<<dim3(64, 5, 4), blk, 0, stream>>>(Wq2, 288, B3, XB288, B0, XB288, 288, 288, nullptr);
  k_gemm<<<dim3(64, 5, 4), blk, 0, stream>>>(Wk2, 288, B3, XB288, B1, XB288, 288, 288, nullptr);
  k_gemm<<<dim3(64, 5, 4), blk, 0, stream>>>(Wv2, 288, B3, XB288, B2, XB288, 288, 288, nullptr);
  // flash self-attention (+gamma residual with cat)
  k_flash<288><<<dim3(64, 1, 4), blk, 0, stream>>>(B0, B1, B2, B3, B4, sa_g);
  // conv3x3 #2 + BN + ReLU
  k_conv3<<<dim3(64, 4, 4), blk, 0, stream>>>(Wp2, B4, XB288, 288, B5);
  k_bnstats<<<256, 256, 0, stream>>>(B5, st + 512, st + 768);
  k_bnrelu<<<16384, 256, 0, stream>>>(B5, st + 512, st + 768, bn2g, bn2b);
  // prediction head
  k_pred<<<64, 256, 0, stream>>>(B5, predw, predb, out);
}
